// MLP_50903952392412
// MI455X (gfx1250) — matrix operand order repaired and checked by reading, not compiled, not run on hardware
//
#include <hip/hip_runtime.h>
#include <hip/hip_bf16.h>

// ---------------------------------------------------------------------------
// NF4 SwiGLU MLP for MI455X (gfx1250, wave32, WMMA).
//   h   = silu(x @ Wg^T) * (x @ Wu^T)      (Wg/Wu dequantized NF4, [M,HD])
//   out = h @ Wd^T                          (Wd dequantized NF4,   [HD,M])
// K-tile == 64 == NF4 blocksize -> one absmax scalar per (row, ktile).
// NF4 codebook lookup via ds_bpermute_b32 (table in lanes 0..15).
// Double-buffered LDS pipeline; v_wmma_f32_16x16x32_f16 compute with 32x32
// register tiles per wave (2x2 accumulators -> 1.5 ds_load_b128 per WMMA);
// async global->LDS (b128) for the f16 activation tile.
// ---------------------------------------------------------------------------

typedef _Float16 v16h __attribute__((ext_vector_type(16)));
typedef float    v8f  __attribute__((ext_vector_type(8)));

#define HD    4096
#define MDIM  14336
#define TOK   4096          // 2 * 2048 tokens
#define NB_HD (HD / 64)     // absmax blocks per gate/up row
#define NB_M  (MDIM / 64)   // absmax blocks per down row
#define NK1   (HD / 64)
#define NK2   (MDIM / 64)

#if __has_builtin(__builtin_amdgcn_global_load_async_to_lds_b128)
#define HAVE_ASYNC_COPY 1
// Builtin signature (from hipcc diagnostic): pointers to vector_size(16) int,
// in global (AS1) / LDS (AS3) address spaces.
typedef int v4i __attribute__((vector_size(16)));
typedef __attribute__((address_space(1))) v4i* g4p_t;
typedef __attribute__((address_space(3))) v4i* l4p_t;
#endif

__device__ __constant__ float NF4_TAB_C[16] = {
    -1.0f, -0.6961928009986877f, -0.5250730514526367f, -0.39491748809814453f,
    -0.28444138169288635f, -0.18477343022823334f, -0.09105003625154495f, 0.0f,
    0.07958029955625534f, 0.16093020141124725f, 0.24611230194568634f,
    0.33791524171829224f, 0.44070982933044434f, 0.5626170039176941f,
    0.7229568362236023f, 1.0f};

// Codebook lookup across lanes: lane l (l<16) holds NF4_TAB_C[l]; bpermute
// gathers value from lane q. EXEC is all-1s everywhere in these kernels.
__device__ __forceinline__ float nf4_lookup(float tabv, int q) {
    return __int_as_float(
        __builtin_amdgcn_ds_bpermute((q & 15) << 2, __float_as_int(tabv)));
}

__device__ __forceinline__ void wait_async() {
#ifdef HAVE_ASYNC_COPY
#if __has_builtin(__builtin_amdgcn_s_wait_asynccnt)
    __builtin_amdgcn_s_wait_asynccnt(0);
#else
    asm volatile("s_wait_asynccnt 0x0" ::: "memory");
#endif
#endif
}

union Pack8 {
    _Float16 h[8];
    uint4    v;
};

// ---- WMMA fragment loads from LDS (ISA 7.12.2 layouts, wave32) -------------
// A (16x32 f16, row-major source, stride lda halves):
//   lane row = lane&15 ; half h: vgpr v=h/2, K = (v/4)*16 + (lane>=16?8:0)
//                                            + 2*(v%4) + (h&1)
__device__ __forceinline__ v16h load_a_frag(const _Float16* p, int lane, int lda) {
    const _Float16* r = p + (lane & 15) * lda;
    const int hi8 = (lane >> 4) << 3;
    v16h a;
#pragma unroll
    for (int g = 0; g < 2; ++g)
#pragma unroll
        for (int v = 0; v < 4; ++v) {
            const int k = g * 16 + hi8 + 2 * v;
            a[g * 8 + 2 * v]     = r[k];
            a[g * 8 + 2 * v + 1] = r[k + 1];
        }
    return a;
}
// B (32x16 f16). Source stored [n][k] row-major (stride ldb halves):
//   lane col n = lane&15 ; K = (h/8)*16 + (lane>=16?8:0) + h%8 (same as A).
__device__ __forceinline__ v16h load_b_frag(const _Float16* p, int lane, int ldb) {
    const _Float16* r = p + (lane & 15) * ldb + ((lane >> 4) << 3);
    v16h b;
#pragma unroll
    for (int i = 0; i < 16; ++i) b[i] = r[(i >> 3) * 16 + (i & 7)];
    return b;
}

__device__ __forceinline__ v8f wmma_f16(v16h a, v16h b, v8f c) {
    return __builtin_amdgcn_wmma_f32_16x16x32_f16(false, a, false, b,
                                                  (short)0, c, false, false);
}

// ---------------------------------------------------------------------------
// Kernel 1: h[t, m] = silu(x Wg^T) * (x Wu^T), f16 into ws.
// Grid: (MDIM/64, TOK/64). Block: 128 threads = 4 waves.
// Wave w: 32x32 output tile: tokens (w&1)*32 + {0,16}, feats (w>>1)*32+{0,16}.
// ---------------------------------------------------------------------------
__global__ __launch_bounds__(128) void nf4_gateup_kernel(
    const float* __restrict__ x, const int* __restrict__ gq,
    const float* __restrict__ gam, const int* __restrict__ uq,
    const float* __restrict__ uam, _Float16* __restrict__ hws) {
    __shared__ _Float16 s_x[2][64 * 64];  // x tile   [tok][k]
    __shared__ _Float16 s_g[2][64 * 64];  // gate W   [m][k] (B layout [n][k])
    __shared__ _Float16 s_u[2][64 * 64];  // up W     [m][k]

    const int tid  = threadIdx.x;
    const int lane = tid & 31;
    const int wid  = tid >> 5;
    const int m0   = blockIdx.x * 64;
    const int t0   = blockIdx.y * 64;
    const int tRow = (wid & 1) << 5;   // token subtile base (+0,+16)
    const int fn   = (wid >> 1) << 5;  // feature subtile base (+0,+16)

    const float tabv = NF4_TAB_C[lane & 15];

    // Stage one 64x64 K-tile: 8 elements (one oct) per thread-iteration,
    // b128 global loads, ds_bpermute dequant, b128 LDS stores.
    auto stage = [&](int kt, int buf) {
#pragma unroll
        for (int i = 0; i < 4; ++i) {
            const int idx8 = tid + i * 128;   // oct index 0..511
            const int rr   = idx8 >> 3;       // tile row 0..63
            const int kk   = (idx8 & 7) << 3; // k offset 0..56
            const int k0   = kt << 6;
            const size_t xb = (size_t)(t0 + rr) * HD + k0 + kk;
            const size_t wb = (size_t)(m0 + rr) * HD + k0 + kk;
            const float4 xa = *reinterpret_cast<const float4*>(&x[xb]);
            const float4 xc = *reinterpret_cast<const float4*>(&x[xb + 4]);
            const int4 qga = *reinterpret_cast<const int4*>(&gq[wb]);
            const int4 qgb = *reinterpret_cast<const int4*>(&gq[wb + 4]);
            const int4 qua = *reinterpret_cast<const int4*>(&uq[wb]);
            const int4 qub = *reinterpret_cast<const int4*>(&uq[wb + 4]);
            const float ag = gam[(m0 + rr) * NB_HD + kt];
            const float au = uam[(m0 + rr) * NB_HD + kt];
            Pack8 px, pg, pu;
            px.h[0] = (_Float16)xa.x; px.h[1] = (_Float16)xa.y;
            px.h[2] = (_Float16)xa.z; px.h[3] = (_Float16)xa.w;
            px.h[4] = (_Float16)xc.x; px.h[5] = (_Float16)xc.y;
            px.h[6] = (_Float16)xc.z; px.h[7] = (_Float16)xc.w;
            pg.h[0] = (_Float16)(nf4_lookup(tabv, qga.x) * ag);
            pg.h[1] = (_Float16)(nf4_lookup(tabv, qga.y) * ag);
            pg.h[2] = (_Float16)(nf4_lookup(tabv, qga.z) * ag);
            pg.h[3] = (_Float16)(nf4_lookup(tabv, qga.w) * ag);
            pg.h[4] = (_Float16)(nf4_lookup(tabv, qgb.x) * ag);
            pg.h[5] = (_Float16)(nf4_lookup(tabv, qgb.y) * ag);
            pg.h[6] = (_Float16)(nf4_lookup(tabv, qgb.z) * ag);
            pg.h[7] = (_Float16)(nf4_lookup(tabv, qgb.w) * ag);
            pu.h[0] = (_Float16)(nf4_lookup(tabv, qua.x) * au);
            pu.h[1] = (_Float16)(nf4_lookup(tabv, qua.y) * au);
            pu.h[2] = (_Float16)(nf4_lookup(tabv, qua.z) * au);
            pu.h[3] = (_Float16)(nf4_lookup(tabv, qua.w) * au);
            pu.h[4] = (_Float16)(nf4_lookup(tabv, qub.x) * au);
            pu.h[5] = (_Float16)(nf4_lookup(tabv, qub.y) * au);
            pu.h[6] = (_Float16)(nf4_lookup(tabv, qub.z) * au);
            pu.h[7] = (_Float16)(nf4_lookup(tabv, qub.w) * au);
            *reinterpret_cast<uint4*>(&s_x[buf][idx8 << 3]) = px.v;
            *reinterpret_cast<uint4*>(&s_g[buf][idx8 << 3]) = pg.v;
            *reinterpret_cast<uint4*>(&s_u[buf][idx8 << 3]) = pu.v;
        }
    };

    v8f cg[2][2] = {}, cu[2][2] = {};

    stage(0, 0);
    for (int kt = 0; kt < NK1; ++kt) {
        __syncthreads();
        const int buf = kt & 1;
        if (kt + 1 < NK1) stage(kt + 1, buf ^ 1);  // overlaps with WMMAs below
#pragma unroll
        for (int kk32 = 0; kk32 < 64; kk32 += 32) {
            v16h a[2], bg[2], bu[2];
            a[0]  = load_a_frag(&s_x[buf][tRow * 64 + kk32], lane, 64);
            a[1]  = load_a_frag(&s_x[buf][(tRow + 16) * 64 + kk32], lane, 64);
            bg[0] = load_b_frag(&s_g[buf][fn * 64 + kk32], lane, 64);
            bg[1] = load_b_frag(&s_g[buf][(fn + 16) * 64 + kk32], lane, 64);
            bu[0] = load_b_frag(&s_u[buf][fn * 64 + kk32], lane, 64);
            bu[1] = load_b_frag(&s_u[buf][(fn + 16) * 64 + kk32], lane, 64);
#pragma unroll
            for (int ti = 0; ti < 2; ++ti)
#pragma unroll
                for (int fi = 0; fi < 2; ++fi) {
                    cg[ti][fi] = wmma_f16(a[ti], bg[fi], cg[ti][fi]);
                    cu[ti][fi] = wmma_f16(a[ti], bu[fi], cu[ti][fi]);
                }
        }
    }

    // SwiGLU epilogue. C layout: lane col n = lane&15; vgpr r -> row
    // m = (lane>=16 ? 8 : 0) + r.  sigmoid via v_rcp_f32 (no IEEE div expansion).
    const int rbase = tRow + ((lane >> 4) << 3);
    const int col0  = m0 + fn + (lane & 15);
#pragma unroll
    for (int ti = 0; ti < 2; ++ti)
#pragma unroll
        for (int fi = 0; fi < 2; ++fi)
#pragma unroll
            for (int r = 0; r < 8; ++r) {
                const int trow = t0 + rbase + ti * 16 + r;
                const float g  = cg[ti][fi][r];
                const float h  =
                    g * __builtin_amdgcn_rcpf(1.0f + __expf(-g)) * cu[ti][fi][r];
                hws[(size_t)trow * MDIM + col0 + fi * 16] = (_Float16)h;
            }
}

// ---------------------------------------------------------------------------
// Kernel 2: out[t, hd] = h @ Wd^T (Wd = dequant(down_q [HD, M])).
// Grid: (HD/64, TOK/64). Block: 128 threads = 4 waves, 32x32 tile per wave.
// h tile is already f16 -> async global->LDS copy (ASYNCcnt).
// ---------------------------------------------------------------------------
__global__ __launch_bounds__(128) void nf4_down_kernel(
    const _Float16* __restrict__ hws, const int* __restrict__ dq,
    const float* __restrict__ dam, float* __restrict__ out) {
    __shared__ _Float16 s_h[2][64 * 64];  // h tile  [tok][m]
    __shared__ _Float16 s_w[2][64 * 64];  // down W  [hd][m] (B layout [n][k])

    const int tid  = threadIdx.x;
    const int lane = tid & 31;
    const int wid  = tid >> 5;
    const int h0   = blockIdx.x * 64;  // HD tile
    const int t0   = blockIdx.y * 64;  // token tile
    const int tRow = (wid & 1) << 5;
    const int hn   = (wid >> 1) << 5;

    const float tabv = NF4_TAB_C[lane & 15];

    auto stage = [&](int kt, int buf) {
        const int k0 = kt << 6;
        {   // h tile: 64 bytes per thread, async b128 copies into LDS
            const int rr = tid >> 1;
            const int kk = (tid & 1) << 5;
            const size_t gsrc = (size_t)(t0 + rr) * MDIM + k0 + kk;
            const int dst = rr * 64 + kk;
#ifdef HAVE_ASYNC_COPY
#pragma unroll
            for (int j = 0; j < 4; ++j)
                __builtin_amdgcn_global_load_async_to_lds_b128(
                    (g4p_t)(hws + gsrc + j * 8), (l4p_t)&s_h[buf][dst + j * 8],
                    0, 0);
#else
#pragma unroll
            for (int j = 0; j < 4; ++j)
                *reinterpret_cast<uint4*>(&s_h[buf][dst + j * 8]) =
                    *reinterpret_cast<const uint4*>(&hws[gsrc + j * 8]);
#endif
        }
#pragma unroll
        for (int i = 0; i < 4; ++i) {   // down weights: oct dequant per thread
            const int idx8 = tid + i * 128;
            const int rr   = idx8 >> 3;
            const int kk   = (idx8 & 7) << 3;
            const size_t wb = (size_t)(h0 + rr) * MDIM + k0 + kk;
            const int4 qa = *reinterpret_cast<const int4*>(&dq[wb]);
            const int4 qb = *reinterpret_cast<const int4*>(&dq[wb + 4]);
            const float am = dam[(h0 + rr) * NB_M + kt];
            Pack8 pw;
            pw.h[0] = (_Float16)(nf4_lookup(tabv, qa.x) * am);
            pw.h[1] = (_Float16)(nf4_lookup(tabv, qa.y) * am);
            pw.h[2] = (_Float16)(nf4_lookup(tabv, qa.z) * am);
            pw.h[3] = (_Float16)(nf4_lookup(tabv, qa.w) * am);
            pw.h[4] = (_Float16)(nf4_lookup(tabv, qb.x) * am);
            pw.h[5] = (_Float16)(nf4_lookup(tabv, qb.y) * am);
            pw.h[6] = (_Float16)(nf4_lookup(tabv, qb.z) * am);
            pw.h[7] = (_Float16)(nf4_lookup(tabv, qb.w) * am);
            *reinterpret_cast<uint4*>(&s_w[buf][idx8 << 3]) = pw.v;
        }
    };

    v8f c[2][2] = {};

    stage(0, 0);
    for (int kt = 0; kt < NK2; ++kt) {
        wait_async();        // async copies for buf (kt&1) must have landed
        __syncthreads();
        const int buf = kt & 1;
        if (kt + 1 < NK2) stage(kt + 1, buf ^ 1);
#pragma unroll
        for (int kk32 = 0; kk32 < 64; kk32 += 32) {
            v16h a[2], b[2];
            a[0] = load_a_frag(&s_h[buf][tRow * 64 + kk32], lane, 64);
            a[1] = load_a_frag(&s_h[buf][(tRow + 16) * 64 + kk32], lane, 64);
            b[0] = load_b_frag(&s_w[buf][hn * 64 + kk32], lane, 64);
            b[1] = load_b_frag(&s_w[buf][(hn + 16) * 64 + kk32], lane, 64);
#pragma unroll
            for (int ti = 0; ti < 2; ++ti)
#pragma unroll
                for (int fi = 0; fi < 2; ++fi)
                    c[ti][fi] = wmma_f16(a[ti], b[fi], c[ti][fi]);
        }
    }

    const int rbase = tRow + ((lane >> 4) << 3);
    const int col0  = h0 + hn + (lane & 15);
#pragma unroll
    for (int ti = 0; ti < 2; ++ti)
#pragma unroll
        for (int fi = 0; fi < 2; ++fi)
#pragma unroll
            for (int r = 0; r < 8; ++r) {
                const int trow = t0 + rbase + ti * 16 + r;
                out[(size_t)trow * HD + col0 + fi * 16] = c[ti][fi][r];
            }
}

extern "C" void kernel_launch(void* const* d_in, const int* in_sizes, int n_in,
                              void* d_out, int out_size, void* d_ws, size_t ws_size,
                              hipStream_t stream) {
    (void)in_sizes; (void)n_in; (void)out_size; (void)ws_size;
    const float* x   = (const float*)d_in[0];
    const int*   gq  = (const int*)d_in[1];
    const float* gam = (const float*)d_in[2];
    const int*   uq  = (const int*)d_in[3];
    const float* uam = (const float*)d_in[4];
    const int*   dq  = (const int*)d_in[5];
    const float* dam = (const float*)d_in[6];
    float*       out = (float*)d_out;
    _Float16*    hws = (_Float16*)d_ws;  // TOK x MDIM f16 activation buffer

    dim3 g1(MDIM / 64, TOK / 64);
    nf4_gateup_kernel<<<g1, 128, 0, stream>>>(x, gq, gam, uq, uam, hws);
    dim3 g2(HD / 64, TOK / 64);
    nf4_down_kernel<<<g2, 128, 0, stream>>>(hws, dq, dam, out);
}
